// HybridMemoryMultiFocalPercent_8186207666550
// MI455X (gfx1250) — matrix operand order repaired and checked by reading, not compiled, not run on hardware
//
#include <hip/hip_runtime.h>
#include <hip/hip_bf16.h>
#include <math.h>

// ---------------------------------------------------------------------------
// Problem constants (from reference)
// ---------------------------------------------------------------------------
#define BATCH 256        // B
#define DIM   1024       // D
#define NFEAT 65536      // N
#define NCLUS 8192       // C
#define INV_TEMP 20.0f   // 1/0.05
#define TOPP 0.1f
#define EPSF 1e-6f

typedef __bf16 bf16;
typedef bf16  v16bf __attribute__((ext_vector_type(16)));
typedef bf16  v8bf  __attribute__((ext_vector_type(8)));
typedef bf16  v4bf  __attribute__((ext_vector_type(4)));
typedef float v8f   __attribute__((ext_vector_type(8)));
typedef int   g4i   __attribute__((vector_size(16)));   // matches builtin param

#define AS1 __attribute__((address_space(1)))
#define AS3 __attribute__((address_space(3)))

#if defined(__has_builtin)
#if __has_builtin(__builtin_amdgcn_global_load_async_to_lds_b128)
#define HAVE_ASYNC_LDS 1
#endif
#endif

__device__ __forceinline__ void wait_async_lds() {
#if defined(HAVE_ASYNC_LDS)
#if __has_builtin(__builtin_amdgcn_s_wait_asynccnt)
    __builtin_amdgcn_s_wait_asynccnt(0);
#else
    asm volatile("s_wait_asynccnt 0x0" ::: "memory");
#endif
#endif
}

// ---------------------------------------------------------------------------
// K0: zero a float range (simC + nums)
// ---------------------------------------------------------------------------
__global__ void k_zero(float* __restrict__ p, int n) {
    int i = blockIdx.x * blockDim.x + threadIdx.x;
    int stride = gridDim.x * blockDim.x;
    for (; i < n; i += stride) p[i] = 0.0f;
}

// ---------------------------------------------------------------------------
// K1: L2-normalize each row of results and convert to bf16
// ---------------------------------------------------------------------------
__global__ void k_normalize(const float* __restrict__ results,
                            bf16* __restrict__ inbf) {
    __shared__ float red[256];
    int row = blockIdx.x;
    int tid = threadIdx.x;
    const float4 v = *(const float4*)(results + (size_t)row * DIM + tid * 4);
    float ss = v.x * v.x + v.y * v.y + v.z * v.z + v.w * v.w;
    red[tid] = ss;
    __syncthreads();
    for (int s = 128; s > 0; s >>= 1) {
        if (tid < s) red[tid] += red[tid + s];
        __syncthreads();
    }
    float inv = rsqrtf(red[0]);
    v4bf o;
    o[0] = (bf16)(v.x * inv); o[1] = (bf16)(v.y * inv);
    o[2] = (bf16)(v.z * inv); o[3] = (bf16)(v.w * inv);
    *(v4bf*)(inbf + (size_t)row * DIM + tid * 4) = o;
}

// ---------------------------------------------------------------------------
// K2: cluster member counts
// ---------------------------------------------------------------------------
__global__ void k_counts(const int* __restrict__ labels, float* __restrict__ nums) {
    int i = blockIdx.x * blockDim.x + threadIdx.x;
    if (i < NFEAT) atomicAdd(&nums[labels[i]], 1.0f);
}

// ---------------------------------------------------------------------------
// K3: WMMA GEMM  sims[b,n] = <inputs_bf16[b,:], features[n,:]> / TEMP
//     fused segment-sum epilogue: simC[label[n]*B + b] += sims[b,n]
//
// Block: 256 threads = 8 waves; block tile M=64 x N=128, K-chunk 64.
// Wave tile: 16(M) x 64(N) = 4 accumulators; A fragment reused 4x.
// A (bf16, precomputed) staged via async global->LDS copies (ASYNCcnt path).
// B (features fp32) read from HBM once, cvt to bf16 in-register, ds_store'd.
// Grid has batch tiles innermost so the 4 consumers of each feature tile are
// launch-adjacent -> features stream from HBM exactly once (~256MB @ 23TB/s).
// ---------------------------------------------------------------------------
#define BM 64
#define BN 128
#define BKC 64
#define LDT 72   // 64 + 8 bf16 pad -> 144-byte row stride

__global__ __launch_bounds__(256)
void k_gemm(const bf16* __restrict__ inbf,
            const float* __restrict__ feat,
            const int* __restrict__ labels,
            float* __restrict__ simC) {
    __shared__ bf16 As[2][BM * LDT];   // 18 KB
    __shared__ bf16 Bs[2][BN * LDT];   // 36 KB

    const int tid  = threadIdx.x;
    const int lane = tid & 31;
    const int w    = tid >> 5;
    const int wm   = w & 3;    // 0..3 -> M sub-tile (16 rows)
    const int wn   = w >> 2;   // 0..1 -> N slot (64 cols)
    const int m0   = blockIdx.x * BM;  // batch tiles innermost (L2 reuse)
    const int n0   = blockIdx.y * BN;

    // ---- staging -----------------------------------------------------------
    auto stageA = [&](int buf, int k0) {
        // 64 rows x 64 bf16 = 8KB -> 2 x 16B per thread
        #pragma unroll
        for (int p = 0; p < 2; ++p) {
            int idx = p * 256 + tid;
            int row = idx >> 3;           // 0..63
            int col = (idx & 7) * 8;      // 0..56
#if defined(HAVE_ASYNC_LDS)
            __builtin_amdgcn_global_load_async_to_lds_b128(
                (AS1 g4i*)(inbf + (size_t)(m0 + row) * DIM + k0 + col),
                (AS3 g4i*)(&As[buf][row * LDT + col]), 0, 0);
#else
            *(v8bf*)&As[buf][row * LDT + col] =
                *(const v8bf*)(inbf + (size_t)(m0 + row) * DIM + k0 + col);
#endif
        }
    };
    auto stageB = [&](int buf, int k0) {
        // 128 rows x 64 fp32 -> bf16; 8 float4 per thread
        int col = (tid & 15) * 4;         // 0..60
        int rb  = tid >> 4;               // 0..15
        #pragma unroll
        for (int r = 0; r < 8; ++r) {
            int row = r * 16 + rb;        // 0..127
            const float4 f = *(const float4*)(feat + (size_t)(n0 + row) * DIM + k0 + col);
            v4bf o;
            o[0] = (bf16)f.x; o[1] = (bf16)f.y; o[2] = (bf16)f.z; o[3] = (bf16)f.w;
            *(v4bf*)&Bs[buf][row * LDT + col] = o;
        }
    };

    v8f acc0 = {}, acc1 = {}, acc2 = {}, acc3 = {};

    stageA(0, 0);
    stageB(0, 0);
    wait_async_lds();
    __syncthreads();

    const int half = lane >> 4;           // lane half selects K sub-range
    const int mr   = lane & 15;

    for (int kc = 0; kc < DIM / BKC; ++kc) {
        const int cur = kc & 1;
        if (kc + 1 < DIM / BKC) {
            stageA(cur ^ 1, (kc + 1) * BKC);
            stageB(cur ^ 1, (kc + 1) * BKC);
        }
        const bf16* abase = &As[cur][(wm * 16 + mr) * LDT];
        const bf16* bbase = &Bs[cur][(wn * 64 + mr) * LDT];
        #pragma unroll
        for (int ks = 0; ks < BKC; ks += 32) {
            // A 16x32 bf16: lanes 0-15 K={0..7,16..23}, lanes 16-31 +8
            v8bf alo = *(const v8bf*)(abase + ks + half * 8);
            v8bf ahi = *(const v8bf*)(abase + ks + 16 + half * 8);
            v16bf a;
            #pragma unroll
            for (int i = 0; i < 8; ++i) { a[i] = alo[i]; a[i + 8] = ahi[i]; }
            // 4 N sub-tiles share the A fragment
            #pragma unroll
            for (int t = 0; t < 4; ++t) {
                const bf16* bp = bbase + t * 16 * LDT + ks + half * 8;
                v8bf blo = *(const v8bf*)(bp);
                v8bf bhi = *(const v8bf*)(bp + 16);
                v16bf b;
                #pragma unroll
                for (int i = 0; i < 8; ++i) { b[i] = blo[i]; b[i + 8] = bhi[i]; }
                v8f& acc = (t == 0) ? acc0 : (t == 1) ? acc1 : (t == 2) ? acc2 : acc3;
                acc = __builtin_amdgcn_wmma_f32_16x16x32_bf16(
                    false, a, false, b, (short)0, acc, false, false);
            }
        }
        wait_async_lds();
        __syncthreads();
    }

    // Epilogue: lane covers N = wn*64 + t*16 + (lane&15),
    //           VGPR j covers M = wm*16 + half*8 + j
    #pragma unroll
    for (int t = 0; t < 4; ++t) {
        const v8f& acc = (t == 0) ? acc0 : (t == 1) ? acc1 : (t == 2) ? acc2 : acc3;
        const int gn = n0 + wn * 64 + t * 16 + mr;
        const int c  = labels[gn];
        float* dst = simC + (size_t)c * BATCH + m0 + wm * 16 + half * 8;
        #pragma unroll
        for (int j = 0; j < 8; ++j)
            atomicAdd(dst + j, acc[j] * INV_TEMP);
    }
}

// ---------------------------------------------------------------------------
// K4: per-row multi-focal loss.  One 1024-thread block per batch row.
// ---------------------------------------------------------------------------
__global__ __launch_bounds__(1024)
void k_focal(const float* __restrict__ simC,
             const float* __restrict__ nums,
             const int* __restrict__ labels,
             const int* __restrict__ indexes,
             float* __restrict__ lossPart) {
    __shared__ float s[NCLUS];
    __shared__ float aux[1024];
    __shared__ int   auxi[1024];
    __shared__ float sh_et;

    const int b   = blockIdx.x;
    const int tid = threadIdx.x;
    const int t   = labels[indexes[b]];

    // fill: e[c] = exp(simC[c,b]/nums[c]) * mask, zeroed at the positive
    #pragma unroll
    for (int e = 0; e < 8; ++e) {
        int c = tid * 8 + e;
        float cnt = nums[c];
        float ev = 0.0f;
        if (cnt > 0.0f) ev = expf(simC[(size_t)c * BATCH + b] / cnt);
        float entry = ev;
        if (c == t) { sh_et = ev; entry = 0.0f; }
        s[c] = entry;
    }
    __syncthreads();

    // bitonic sort, descending
    for (int k = 2; k <= NCLUS; k <<= 1) {
        for (int j = k >> 1; j > 0; j >>= 1) {
            #pragma unroll
            for (int e = 0; e < 8; ++e) {
                int i = tid + e * 1024;
                int ixj = i ^ j;
                if (ixj > i) {
                    float a = s[i], bb = s[ixj];
                    bool desc = ((i & k) == 0);
                    if (desc ? (a < bb) : (a > bb)) { s[i] = bb; s[ixj] = a; }
                }
            }
            __syncthreads();
        }
    }

    // block scan over sorted values (8 contiguous per thread + Hillis-Steele)
    const int base = tid * 8;
    float lsum = 0.0f;
    #pragma unroll
    for (int e = 0; e < 8; ++e) lsum += s[base + e];
    aux[tid] = lsum;
    __syncthreads();
    for (int off = 1; off < 1024; off <<= 1) {
        float v = (tid >= off) ? aux[tid - off] : 0.0f;
        __syncthreads();
        aux[tid] += v;
        __syncthreads();
    }
    const float total  = aux[1023];          // neg_sum
    const float excl   = aux[tid] - lsum;
    const float target = TOPP * total;

    // argmin |cumsum - target| (first occurrence on ties)
    float best = 3.4e38f; int bidx = 0;
    float run = excl;
    #pragma unroll
    for (int e = 0; e < 8; ++e) {
        run += s[base + e];
        float d = fabsf(run - target);
        if (d < best) { best = d; bidx = base + e; }
    }
    __syncthreads();
    aux[tid] = best; auxi[tid] = bidx;
    __syncthreads();
    for (int st = 512; st > 0; st >>= 1) {
        if (tid < st) {
            float ov = aux[tid + st]; int oi = auxi[tid + st];
            if (ov < aux[tid] || (ov == aux[tid] && oi < auxi[tid])) {
                aux[tid] = ov; auxi[tid] = oi;
            }
        }
        __syncthreads();
    }
    const float thr = s[auxi[0]];
    __syncthreads();

    // kept-negative sum: keep ori_neg >= threshold
    float ks = 0.0f;
    #pragma unroll
    for (int e = 0; e < 8; ++e) {
        float v = s[base + e];
        if (v >= thr) ks += v;
    }
    aux[tid] = ks;
    __syncthreads();
    for (int st = 512; st > 0; st >>= 1) {
        if (tid < st) aux[tid] += aux[tid + st];
        __syncthreads();
    }
    if (tid == 0) {
        float et = sh_et;
        float denom = et + aux[0];
        float p = et / (denom + EPSF);
        lossPart[b] = -logf(p + EPSF);
    }
}

// ---------------------------------------------------------------------------
// K5: mean of per-row losses -> scalar output
// ---------------------------------------------------------------------------
__global__ void k_mean(const float* __restrict__ lossPart, float* __restrict__ out) {
    __shared__ float red[256];
    int tid = threadIdx.x;
    red[tid] = lossPart[tid];
    __syncthreads();
    for (int s = 128; s > 0; s >>= 1) {
        if (tid < s) red[tid] += red[tid + s];
        __syncthreads();
    }
    if (tid == 0) out[0] = red[0] / (float)BATCH;
}

// ---------------------------------------------------------------------------
// Launcher
// ws layout: simC (8192*256 f32) | nums (8192 f32) | inputs_bf16 (256*1024) |
//            lossPart (256 f32)   (~9 MB total)
// ---------------------------------------------------------------------------
extern "C" void kernel_launch(void* const* d_in, const int* in_sizes, int n_in,
                              void* d_out, int out_size, void* d_ws, size_t ws_size,
                              hipStream_t stream) {
    const float* results  = (const float*)d_in[0];
    const float* features = (const float*)d_in[1];
    const int*   indexes  = (const int*)d_in[2];
    const int*   labels   = (const int*)d_in[3];
    float*       out      = (float*)d_out;

    char* ws = (char*)d_ws;
    float* simC     = (float*)ws;                                    // 8 MB
    float* nums     = (float*)(ws + (size_t)NCLUS * BATCH * 4);      // 32 KB
    bf16*  inbf     = (bf16*) (ws + (size_t)NCLUS * BATCH * 4 + NCLUS * 4);
    float* lossPart = (float*)(ws + (size_t)NCLUS * BATCH * 4 + NCLUS * 4
                                  + (size_t)BATCH * DIM * 2);

    const int nz = NCLUS * BATCH + NCLUS;
    k_zero<<<1024, 256, 0, stream>>>(simC, nz);

    k_normalize<<<BATCH, 256, 0, stream>>>(results, inbf);
    k_counts<<<NFEAT / 256, 256, 0, stream>>>(labels, nums);

    dim3 gg(BATCH / BM, NFEAT / BN);   // batch tiles innermost: 4 x 512
    k_gemm<<<gg, 256, 0, stream>>>(inbf, features, labels, simC);

    k_focal<<<BATCH, 1024, 0, stream>>>(simC, nums, labels, indexes, lossPart);
    k_mean<<<1, 256, 0, stream>>>(lossPart, out);
}
